// LSTMUnit_24704651886898
// MI455X (gfx1250) — matrix operand order repaired and checked by reading, not compiled, not run on hardware
//
#include <hip/hip_runtime.h>
#include <hip/hip_bf16.h>
#include <math.h>

// ---------------------------------------------------------------------------
// LSTM: T=256, B=64, IN=HID=OUT=1024.
//   1) pack weights transposed to bf16 (WMMA B-matrix wants K contiguous per N)
//   2) 256x lstm_step_kernel: gates = x_t@W4x + h@W4h + b  (bf16 WMMA, f32 acc)
//      A-operand (x_t / h) staged into LDS by the Tensor Data Mover (TDM),
//      double buffered on TENSORcnt; fused sigmoid/tanh + c/h update via LDS.
//   3) batched Y = H@Wy + by  (bf16 WMMA)
//   4) copy final h, c
// ---------------------------------------------------------------------------

#define T_STEPS 256
#define BATCH   64
#define HID     1024
#define GATES4  4096
#define BH      (BATCH * HID)          // 65536
#define TB      (T_STEPS * BATCH)      // 16384

typedef __bf16 bf16;
typedef __attribute__((ext_vector_type(16))) __bf16 v16bf;
typedef __attribute__((ext_vector_type(8)))  __bf16 v8bf;
typedef __attribute__((ext_vector_type(8)))  float  v8f;
typedef __attribute__((ext_vector_type(4)))  unsigned int v4u;
typedef __attribute__((ext_vector_type(8)))  int v8i_;
typedef __attribute__((ext_vector_type(4)))  int v4i_;

#if __has_builtin(__builtin_amdgcn_tensor_load_to_lds)
#define USE_TDM 1
#else
#define USE_TDM 0
#endif

// ---- WMMA fragment loaders (CDNA5 16x16x32 bf16 layouts, ISA 7.12.2) ------

// A matrix 16x32 bf16, row-major source (lda elems). Lane l: row = l&15,
// VGPR0..3 hold K = k0+kh .. +7, VGPR4..7 hold K = k0+kh+16 .. +23, kh = 8*(l>=16).
static __device__ inline v16bf load_a_frag(const bf16* A, int lda, int row0,
                                           int k0, int lane) {
    const int r  = row0 + (lane & 15);
    const int kh = (lane >> 4) << 3;
    const bf16* p = A + (size_t)r * lda + k0 + kh;
    v8bf lo = *(const v8bf*)(p);
    v8bf hi = *(const v8bf*)(p + 16);
    return __builtin_shufflevector(lo, hi, 0, 1, 2, 3, 4, 5, 6, 7,
                                           8, 9, 10, 11, 12, 13, 14, 15);
}

// B matrix 32x16 bf16 from transposed weights WT[N][K] (ldb = K stride).
// Lane l: col = l&15, K = k0+kh .. +7 then k0+kh+16 .. +23, kh = 8*(l>=16).
static __device__ inline v16bf load_b_frag(const bf16* WT, int ldb, int n0,
                                           int k0, int lane) {
    const int c  = n0 + (lane & 15);
    const int ks = k0 + ((lane >> 4) << 3);
    const bf16* p = WT + (size_t)c * ldb + ks;
    v8bf lo = *(const v8bf*)(p);
    v8bf hi = *(const v8bf*)(p + 16);
    return __builtin_shufflevector(lo, hi, 0, 1, 2, 3, 4, 5, 6, 7,
                                           8, 9, 10, 11, 12, 13, 14, 15);
}

static __device__ inline v8f wmma_bf16(v16bf a, v16bf b, v8f c) {
    return __builtin_amdgcn_wmma_f32_16x16x32_bf16(
        /*neg_a=*/false, a, /*neg_b=*/false, b,
        /*c_mod=*/(short)0, c, /*reuse_a=*/false, /*reuse_b=*/false);
}

static __device__ inline float sigmoidf_(float x) {
    return 1.0f / (1.0f + __expf(-x));
}
static __device__ inline float tanhf_(float x) {
    return 2.0f / (1.0f + __expf(-2.0f * x)) - 1.0f;
}

// ---- TDM helpers -----------------------------------------------------------

static __device__ inline unsigned lds_offset(const void* p) {
    // generic -> LDS address space; ptrtoint of an AS(3) pointer is the LDS
    // byte offset, which is what the D# lds_addr field wants.
    return (unsigned)(size_t)(__attribute__((address_space(3))) const void*)p;
}

static __device__ inline void wait_tensorcnt0() {
#if __has_builtin(__builtin_amdgcn_s_wait_tensorcnt)
    __builtin_amdgcn_s_wait_tensorcnt((short)0);
#else
    asm volatile("s_wait_tensorcnt 0x0" ::: "memory");
#endif
}

#if USE_TDM
// DMA a 64-row x 256-byte tile (row stride 2048 B) from global into LDS,
// packed contiguously (64 x 128 bf16). D# per CDNA5 ISA ch.8:
//   elements are 8 B (data_size=3): tile 32 x 64, tensor_dim0_stride = 256.
static __device__ inline void tdm_load_tile(const bf16* gsrc, unsigned lds_byte) {
    const unsigned long long ga = (unsigned long long)(size_t)gsrc;
    v4u g0;
    g0[0] = 1u;                                                 // count=1, user mode
    g0[1] = lds_byte;                                           // lds_addr
    g0[2] = (unsigned)ga;                                       // global_addr[31:0]
    g0[3] = (unsigned)((ga >> 32) & 0x01FFFFFFu) | (2u << 30);  // [56:32] | type=2
    const unsigned td0 = 32, td1 = 64, tile0 = 32, tile1 = 64;  // 8-byte units
    const unsigned long long s0 = 256;                          // row stride, 8-byte units
    v8i_ g1;
    g1[0] = (int)(3u << 16);                                    // data_size = 8 B
    g1[1] = (int)((td0 & 0xFFFFu) << 16);                       // tensor_dim0[15:0]
    g1[2] = (int)((td0 >> 16) | ((td1 & 0xFFFFu) << 16));       // td0 hi | td1 lo
    g1[3] = (int)((td1 >> 16) | (tile0 << 16));                 // td1 hi | tile_dim0
    g1[4] = (int)(tile1);                                       // tile_dim1; tile_dim2=0
    g1[5] = (int)(unsigned)(s0 & 0xFFFFFFFFu);                  // dim0_stride lo
    g1[6] = (int)(unsigned)(s0 >> 32);                          // dim0_stride hi | dim1_stride lo
    g1[7] = 0;                                                  // dim1_stride hi
    v4i_ z4 = {};
    v8i_ z8 = {};
    // clang-23 / therock lane: 6-arg form (g0, g1, g2, g3, ext, cpol)
    __builtin_amdgcn_tensor_load_to_lds(g0, g1, z4, z4, z8, 0);
}
#endif

// Fallback staging: 64 rows x 128 bf16, 16 B per thread-iteration.
static __device__ inline void stage_tile(const bf16* gsrc, bf16* dst, int tid) {
    for (int i = tid; i < 1024; i += 256) {
        const int r = i >> 4, s = i & 15;
        *(v8bf*)(dst + r * 128 + s * 8) =
            *(const v8bf*)(gsrc + (size_t)r * HID + s * 8);
    }
}

// ---------------------------------------------------------------------------
// Packing / conversion kernels
// ---------------------------------------------------------------------------

__global__ void cvt_f32_to_bf16_kernel(const float* __restrict__ src,
                                       bf16* __restrict__ dst, size_t n) {
    size_t i = (size_t)blockIdx.x * blockDim.x + threadIdx.x;
    if (i < n) dst[i] = (bf16)src[i];
}

// W fp32 [K=1024][N=1024] row-major  ->  WT bf16 [N][K]
__global__ void pack_wT_kernel(const float* __restrict__ W,
                               bf16* __restrict__ WT) {
    size_t i = (size_t)blockIdx.x * blockDim.x + threadIdx.x;  // n*1024 + k
    if (i < (size_t)HID * HID) {
        int n = (int)(i >> 10);
        int k = (int)(i & 1023);
        WT[i] = (bf16)W[(size_t)k * HID + n];
    }
}

__global__ void pack_bias_kernel(const float* __restrict__ bf_,
                                 const float* __restrict__ bi_,
                                 const float* __restrict__ bc_,
                                 const float* __restrict__ bo_,
                                 float* __restrict__ B4) {
    int i = blockIdx.x * blockDim.x + threadIdx.x;
    if (i < HID) {
        B4[i]           = bf_[i];
        B4[HID + i]     = bi_[i];
        B4[2 * HID + i] = bc_[i];
        B4[3 * HID + i] = bo_[i];
    }
}

__global__ void init_state_kernel(const float* __restrict__ h0,
                                  const float* __restrict__ c0,
                                  bf16* __restrict__ HB,
                                  float* __restrict__ cstate,
                                  float* __restrict__ hstate) {
    int i = blockIdx.x * blockDim.x + threadIdx.x;
    if (i < BH) {
        HB[i]     = (bf16)h0[i];
        hstate[i] = h0[i];
        cstate[i] = c0[i];
    }
}

// ---------------------------------------------------------------------------
// Recurrent step: one workgroup per 16 hidden columns.
// 8 waves = 4 gates x 2 row-halves; each wave computes two 16x16 tiles over
// K = 1024 (x part) + 1024 (h part), consumed as 16 chunks of K=128.
// Each chunk's A-tile (64 rows x 128 bf16 = 16 KB) is DMA'd into LDS by the
// TDM (wave 0 issues, TENSORcnt-waited), double buffered against compute.
// ---------------------------------------------------------------------------
__global__ void __launch_bounds__(256)
lstm_step_kernel(const bf16* __restrict__ XB,     // [T*B][IN] bf16
                 const bf16* __restrict__ WX4T,   // [4H][IN]  bf16 (transposed)
                 const bf16* __restrict__ WH4T,   // [4H][HID] bf16 (transposed)
                 const float* __restrict__ B4,    // [4H]
                 bf16* __restrict__ HB,           // [T+1][B][HID] bf16 h history
                 float* __restrict__ cstate,      // [B][HID]
                 float* __restrict__ hstate,      // [B][HID]
                 int t) {
    __shared__ __align__(16) bf16 abuf[2][BATCH * 128];   // 2 x 16 KB A staging
    __shared__ float gbuf[4][BATCH][16];                  // 16 KB gate exchange

    const int lane = threadIdx.x & 31;
    const int w    = threadIdx.x >> 5;
    const int g    = w & 3;                       // gate: 0=f 1=i 2=cd 3=o
    const int half = w >> 2;                      // row half (0: rows 0..31, 1: 32..63)
    const int hc   = blockIdx.x;                  // hidden-col tile, 0..63
    const int n0g  = g * HID + hc * 16;           // column in 4096-wide gate space
    const int m0   = half * 32;

    const bf16* Ax = XB + (size_t)t * BH;         // x_t rows [B][IN]
    const bf16* Ah = HB + (size_t)t * BH;         // h_{t-1} rows [B][HID]

    v8f acc0 = {};
    v8f acc1 = {};

#if USE_TDM
    const unsigned lds0 = lds_offset(&abuf[0][0]);
    const unsigned lds1 = lds_offset(&abuf[1][0]);
    if (w == 0) tdm_load_tile(Ax, lds0);          // prefetch chunk 0
#endif

    for (int c = 0; c < 16; ++c) {
#if USE_TDM
        if (w == 0) wait_tensorcnt0();            // chunk c landed in abuf[c&1]
        __syncthreads();
        if (c < 15 && w == 0) {
            const int cn = c + 1;
            const bf16* gsrc = (cn < 8) ? (Ax + cn * 128) : (Ah + (cn - 8) * 128);
            tdm_load_tile(gsrc, (cn & 1) ? lds1 : lds0);
        }
#else
        {
            const bf16* gsrc = (c < 8) ? (Ax + c * 128) : (Ah + (c - 8) * 128);
            stage_tile(gsrc, abuf[c & 1], threadIdx.x);
        }
        __syncthreads();
#endif
        const bf16* WT = (c < 8) ? WX4T : WH4T;
        const int kb   = (c & 7) * 128;
        const bf16* As = abuf[c & 1];
#pragma unroll
        for (int kk = 0; kk < 128; kk += 32) {
            v16bf b  = load_b_frag(WT, HID, n0g, kb + kk, lane);
            v16bf a0 = load_a_frag(As, 128, m0, kk, lane);
            v16bf a1 = load_a_frag(As, 128, m0 + 16, kk, lane);
            acc0 = wmma_bf16(a0, b, acc0);
            acc1 = wmma_bf16(a1, b, acc1);
        }
        __syncthreads();   // all waves done reading abuf[c&1] before it is refilled
    }

    // bias + activation, stash into LDS for the cross-gate update
    const int col   = lane & 15;
    const int rbase = (lane >> 4) << 3;
    const float bias = B4[n0g + col];
#pragma unroll
    for (int i = 0; i < 8; ++i) {
        float v0 = acc0[i] + bias;
        float v1 = acc1[i] + bias;
        if (g == 2) { v0 = tanhf_(v0);    v1 = tanhf_(v1);    }
        else        { v0 = sigmoidf_(v0); v1 = sigmoidf_(v1); }
        gbuf[g][m0 + rbase + i][col]      = v0;
        gbuf[g][m0 + 16 + rbase + i][col] = v1;
    }
    __syncthreads();

    // c,h update for this block's 64 rows x 16 hidden cols
    for (int e = threadIdx.x; e < BATCH * 16; e += 256) {
        const int row  = e >> 4;
        const int c16  = e & 15;
        const int gcol = hc * 16 + c16;
        const size_t idx = (size_t)row * HID + gcol;
        const float F  = gbuf[0][row][c16];
        const float I  = gbuf[1][row][c16];
        const float Cd = gbuf[2][row][c16];
        const float O  = gbuf[3][row][c16];
        const float cn = F * cstate[idx] + I * Cd;
        cstate[idx] = cn;
        const float hv = tanhf_(cn) * O;
        hstate[idx] = hv;
        HB[(size_t)(t + 1) * BH + idx] = (bf16)hv;
    }
}

// ---------------------------------------------------------------------------
// Batched output projection: Y[TB][OUT] = H[TB][HID] @ Wy + by
// 8 waves/block; each wave: two 16x16 row tiles sharing one B fragment.
// ---------------------------------------------------------------------------
__global__ void __launch_bounds__(256)
y_gemm_kernel(const bf16* __restrict__ Hall,    // [TB][HID] bf16 (HB slots 1..T)
              const bf16* __restrict__ WYT,     // [OUT][HID] bf16 (transposed)
              const float* __restrict__ by,
              float* __restrict__ Y) {
    const int lane = threadIdx.x & 31;
    const int w    = threadIdx.x >> 5;
    const long task = (long)blockIdx.x * 8 + w;  // (TB/32) * (OUT/16) tasks
    const int m0 = (int)(task >> 6) << 5;        // 512 row-pair tiles
    const int n0 = (int)(task & 63) << 4;        // 64 col tiles

    v8f acc0 = {};
    v8f acc1 = {};
    for (int k0 = 0; k0 < HID; k0 += 32) {
        v16bf b  = load_b_frag(WYT, HID, n0, k0, lane);
        v16bf a0 = load_a_frag(Hall, HID, m0, k0, lane);
        v16bf a1 = load_a_frag(Hall, HID, m0 + 16, k0, lane);
        acc0 = wmma_bf16(a0, b, acc0);
        acc1 = wmma_bf16(a1, b, acc1);
    }

    const int col   = lane & 15;
    const int rbase = (lane >> 4) << 3;
    const float bias = by[n0 + col];
#pragma unroll
    for (int i = 0; i < 8; ++i) {
        Y[(size_t)(m0 + rbase + i) * HID + n0 + col]      = acc0[i] + bias;
        Y[(size_t)(m0 + 16 + rbase + i) * HID + n0 + col] = acc1[i] + bias;
    }
}

__global__ void finalize_kernel(const float* __restrict__ hstate,
                                const float* __restrict__ cstate,
                                float* __restrict__ out_tail) {
    int i = blockIdx.x * blockDim.x + threadIdx.x;
    if (i < BH) {
        out_tail[i]      = hstate[i];
        out_tail[BH + i] = cstate[i];
    }
}

// ---------------------------------------------------------------------------
// Host-side orchestration
// ---------------------------------------------------------------------------
extern "C" void kernel_launch(void* const* d_in, const int* in_sizes, int n_in,
                              void* d_out, int out_size, void* d_ws, size_t ws_size,
                              hipStream_t stream) {
    const float* steps = (const float*)d_in[0];
    const float* h0    = (const float*)d_in[1];
    const float* c0    = (const float*)d_in[2];
    const float* Wfx   = (const float*)d_in[3];
    const float* Wfh   = (const float*)d_in[4];
    const float* bf_   = (const float*)d_in[5];
    const float* Wix   = (const float*)d_in[6];
    const float* Wih   = (const float*)d_in[7];
    const float* bi_   = (const float*)d_in[8];
    const float* Wcx   = (const float*)d_in[9];
    const float* Wch   = (const float*)d_in[10];
    const float* bc_   = (const float*)d_in[11];
    const float* Wox   = (const float*)d_in[12];
    const float* Woh   = (const float*)d_in[13];
    const float* bo_   = (const float*)d_in[14];
    const float* Wy    = (const float*)d_in[15];
    const float* by    = (const float*)d_in[16];

    // workspace carve-up (~82.6 MB total)
    char* p = (char*)d_ws;
    bf16* XB    = (bf16*)p;  p += (size_t)TB * HID * sizeof(bf16);           // 32 MB
    bf16* WX4T  = (bf16*)p;  p += (size_t)GATES4 * HID * sizeof(bf16);       // 8 MB
    bf16* WH4T  = (bf16*)p;  p += (size_t)GATES4 * HID * sizeof(bf16);       // 8 MB
    bf16* WYT   = (bf16*)p;  p += (size_t)HID * HID * sizeof(bf16);          // 2 MB
    float* B4   = (float*)p; p += (size_t)GATES4 * sizeof(float);
    bf16* HB    = (bf16*)p;  p += (size_t)(T_STEPS + 1) * BH * sizeof(bf16); // 32.1 MB
    float* cst  = (float*)p; p += (size_t)BH * sizeof(float);
    float* hst  = (float*)p; p += (size_t)BH * sizeof(float);

    const size_t nX = (size_t)TB * HID;
    cvt_f32_to_bf16_kernel<<<(unsigned)((nX + 255) / 256), 256, 0, stream>>>(steps, XB, nX);

    const unsigned wblk = (HID * HID + 255) / 256;  // 4096 blocks
    pack_wT_kernel<<<wblk, 256, 0, stream>>>(Wfx, WX4T + 0 * (size_t)HID * HID);
    pack_wT_kernel<<<wblk, 256, 0, stream>>>(Wix, WX4T + 1 * (size_t)HID * HID);
    pack_wT_kernel<<<wblk, 256, 0, stream>>>(Wcx, WX4T + 2 * (size_t)HID * HID);
    pack_wT_kernel<<<wblk, 256, 0, stream>>>(Wox, WX4T + 3 * (size_t)HID * HID);
    pack_wT_kernel<<<wblk, 256, 0, stream>>>(Wfh, WH4T + 0 * (size_t)HID * HID);
    pack_wT_kernel<<<wblk, 256, 0, stream>>>(Wih, WH4T + 1 * (size_t)HID * HID);
    pack_wT_kernel<<<wblk, 256, 0, stream>>>(Wch, WH4T + 2 * (size_t)HID * HID);
    pack_wT_kernel<<<wblk, 256, 0, stream>>>(Woh, WH4T + 3 * (size_t)HID * HID);
    pack_wT_kernel<<<wblk, 256, 0, stream>>>(Wy,  WYT);

    pack_bias_kernel<<<4, 256, 0, stream>>>(bf_, bi_, bc_, bo_, B4);
    init_state_kernel<<<BH / 256, 256, 0, stream>>>(h0, c0, HB, cst, hst);

    // sequential scan: 64 blocks x 256 threads per step; 65536 WMMA / step
    for (int t = 0; t < T_STEPS; ++t) {
        lstm_step_kernel<<<64, 256, 0, stream>>>(XB, WX4T, WH4T, B4, HB, cst, hst, t);
    }

    // batched output projection: (TB/32)*(OUT/16)/8 = 4096 blocks
    y_gemm_kernel<<<4096, 256, 0, stream>>>(HB + BH, WYT, by, (float*)d_out);

    finalize_kernel<<<BH / 256, 256, 0, stream>>>(hst, cst,
                                                  (float*)d_out + (size_t)TB * HID);
}
